// MyEGATRegressor_35485019799706
// MI455X (gfx1250) — compile-verified, matrix operand order checked by reading, not run on hardware
//
#include <hip/hip_runtime.h>
#include <hip/hip_bf16.h>
#include <math.h>
#include <stdint.h>

#define NNODES 50000
#define NEDGES 800000
#define NGRAPH 8
#define DIM    128      // NODE_EMBED = EDGE_EMBED = HEADS*OUT_* = 128

typedef __attribute__((ext_vector_type(16))) _Float16 v16h;
typedef __attribute__((ext_vector_type(8)))  float    v8f;

__device__ __forceinline__ float sigm_(float x){ return 1.f/(1.f+__expf(-x)); }
__device__ __forceinline__ float elu_(float x){ return x>0.f ? x : __expf(x)-1.f; }
__device__ __forceinline__ float silu_(float x){ return x*sigm_(x); }

__device__ __forceinline__ void atomicMaxF(float* a, float v){
    if (v >= 0.f) atomicMax((int*)a, __float_as_int(v));
    else          atomicMin((unsigned int*)a, __float_as_uint(v));
}

// ---------------------------------------------------------------- utilities
__global__ void fill_k(float* __restrict__ p, float v, long n){
    long i = (long)blockIdx.x*blockDim.x + threadIdx.x;
    if (i < n) p[i] = v;
}

__global__ void embed_k(const int* __restrict__ types, const float* __restrict__ tab,
                        float* __restrict__ nf, long n){
    long i = (long)blockIdx.x*blockDim.x + threadIdx.x;
    if (i >= n) return;
    int row = (int)(i >> 7), c = (int)(i & 127);
    nf[i] = tab[(size_t)types[row]*DIM + c];
}

__global__ void elu_k(float* __restrict__ p, long n){
    long i = (long)blockIdx.x*blockDim.x + threadIdx.x;
    if (i < n) p[i] = elu_(p[i]);
}

// Pack weight W (128x128 f32, row-major K x N) into WMMA B-fragment lane order:
// Bp[((tn*4 + kt)*32 + lane)*16 + j] = f16( W[k, tn*16 + (lane&15)] ),
// k = kt*32 + (lane>>4)*8 + (j<8 ? j : 16+(j-8)).
// Each lane's 16 f16 fragment becomes one contiguous 32B read in the GEMM.
__global__ void pack_b_k(const float* __restrict__ Wf, _Float16* __restrict__ Bp){
    int idx = blockIdx.x*blockDim.x + threadIdx.x;    // 16384
    if (idx >= 16384) return;
    int j    = idx & 15;
    int lane = (idx >> 4) & 31;
    int kt   = (idx >> 9) & 3;
    int tn   = idx >> 11;
    int half = lane >> 4, l15 = lane & 15;
    int k = kt*32 + half*8 + ((j < 8) ? j : 16 + (j - 8));
    Bp[idx] = (_Float16)Wf[(size_t)k*128 + tn*16 + l15];
}

// ------------------------------------------------------- shared GEMM pieces
// async-stage one 16x128 fp32 A tile (8 KB) into LDS, tracked by ASYNCcnt
__device__ __forceinline__ void stage_A_async(const float* A, int tm, int lda,
                                              float* As, int t){
    int row = t >> 4;                 // 0..15
    int col = (t & 15) << 3;          // 0..120 step 8
    unsigned long long ga =
        (unsigned long long)(uintptr_t)(A + (size_t)(tm + row)*lda + col);
    unsigned lp = (unsigned)(uintptr_t)As + (unsigned)((row*128 + col)*4);
    asm volatile("global_load_async_to_lds_b128 %0, %1, off"
                 :: "v"(lp), "v"(ga) : "memory");
    asm volatile("global_load_async_to_lds_b128 %0, %1, off offset:16"
                 :: "v"(lp), "v"(ga) : "memory");
    asm volatile("s_wait_asynccnt 0x0" ::: "memory");
}

__device__ __forceinline__ v8f wmma_tile_128(const float* As, const _Float16* Bp,
                                             int wave, int lane){
    const int half = lane >> 4, l15 = lane & 15;
    v8f acc = {};
#pragma unroll
    for (int kb = 0; kb < 128; kb += 32) {
        v16h av, bv;
        const float* ap0 = &As[l15*128 + kb + half*8];   // ds_load_b128 x4 + cvt
#pragma unroll
        for (int j = 0; j < 8; ++j) av[j]   = (_Float16)ap0[j];
#pragma unroll
        for (int j = 0; j < 8; ++j) av[8+j] = (_Float16)ap0[16+j];
        bv = ((const v16h*)Bp)[(wave*4 + (kb >> 5))*32 + lane]; // 2x b128
        acc = __builtin_amdgcn_wmma_f32_16x16x32_f16(
                  false, av, false, bv, (short)0, acc, false, false);
    }
    return acc;
}

// ------------------------------------------------------- plain WMMA GEMM
// C[16 rows, 128 cols] = A * Bh.  Block = 8 waves = the 8 N-tiles, so the
// streaming A operand is read from HBM exactly once. RT stores (L2-resident C).
__global__ void __launch_bounds__(256)
gemm128_wmma_f16(const float* __restrict__ A, int lda,
                 const _Float16* __restrict__ Bp,
                 float* __restrict__ C, int ldc, int ccol)
{
    __shared__ float As[16*128];
    const int tm = blockIdx.x << 4;
    const int t  = threadIdx.x;
    stage_A_async(A, tm, lda, As, t);
    __syncthreads();

    const int wave = t >> 5, lane = t & 31;
    const int half = lane >> 4, l15 = lane & 15;
    v8f acc = wmma_tile_128(As, Bp, wave, lane);

    float* Crow = C + (size_t)(tm + half*8)*ldc + ccol + (wave << 4) + l15;
#pragma unroll
    for (int r = 0; r < 8; ++r) Crow[(size_t)r*ldc] = acc[r];
}

// ------------------------------------- fused edge GEMM + EGAT edge epilogue
// One block = 16 edges x full 128 cols. After WMMA:
//   f = lrelu(fproj + ni[src] + nj[dst] + b)   (gathers hit the L2-resident
//   N x 384 projection table), per-head attention logits reduced via
//   __shfl_xor + LDS, and elu(f) written non-temporally (410MB stream that
//   cannot fit the 192MB L2). Eliminates the fproj round-trip entirely.
__global__ void __launch_bounds__(256)
gemm_edge_fused(const float* __restrict__ A,
                const _Float16* __restrict__ Bp,
                const float* __restrict__ nproj,
                const float* __restrict__ bias,
                const float* __restrict__ attn,
                const int* __restrict__ src, const int* __restrict__ dst,
                float* __restrict__ Fout, float* __restrict__ elog)
{
    __shared__ float As[16*128];
    __shared__ int   sE[16], dE[16];
    __shared__ float esh[16*4];
    const int tm = blockIdx.x << 4;
    const int t  = threadIdx.x;
    stage_A_async(A, tm, 128, As, t);
    if (t < 16) { sE[t] = src[tm + t]; dE[t] = dst[tm + t]; }
    if (t >= 192) esh[t - 192] = 0.f;
    __syncthreads();

    const int wave = t >> 5, lane = t & 31;
    const int half = lane >> 4, l15 = lane & 15;
    const int col  = (wave << 4) + l15;
    v8f acc = wmma_tile_128(As, Bp, wave, lane);

    const float bc = bias[col];
    const float ac = attn[col];          // attn (4,32) flat == channel index
    const int head = col >> 5;
#pragma unroll
    for (int r = 0; r < 8; ++r) {
        int row = half*8 + r;
        float f = acc[r] + nproj[(size_t)sE[row]*384 + col]
                         + nproj[(size_t)dE[row]*384 + 128 + col] + bc;
        f = f > 0.f ? f : 0.01f*f;                 // leaky_relu
        float part = f * ac;                       // head-logit contribution
        part += __shfl_xor(part, 1, 16);
        part += __shfl_xor(part, 2, 16);
        part += __shfl_xor(part, 4, 16);
        part += __shfl_xor(part, 8, 16);           // sum over this half's 16 cols
        if (l15 == 0) atomicAdd(&esh[row*4 + head], part);
        __builtin_nontemporal_store(elu_(f), Fout + (size_t)(tm + row)*DIM + col);
    }
    __syncthreads();
    if (t < 64) elog[(size_t)(tm + (t >> 2))*4 + (t & 3)] = esh[t];
}

// ------------------------------------------------------------- EGAT kernels
__global__ void seg_max_k(const float* __restrict__ elog, const int* __restrict__ dst,
                          float* __restrict__ segmax, int nE){
    long i = (long)blockIdx.x*blockDim.x + threadIdx.x;
    if (i >= (long)nE*4) return;
    int e = (int)(i >> 2), h = (int)(i & 3);
    atomicMaxF(&segmax[(size_t)dst[e]*4 + h], elog[i]);
}

__global__ void seg_expsum_k(float* __restrict__ elog, const int* __restrict__ dst,
                             const float* __restrict__ segmax, float* __restrict__ segsum, int nE){
    long i = (long)blockIdx.x*blockDim.x + threadIdx.x;
    if (i >= (long)nE*4) return;
    int e = (int)(i >> 2), h = (int)(i & 3);
    float ex = __expf(elog[i] - segmax[(size_t)dst[e]*4 + h]);
    elog[i] = ex;
    atomicAdd(&segsum[(size_t)dst[e]*4 + h], ex);
}

__global__ void aggregate_k(const float* __restrict__ nproj, const float* __restrict__ elog,
                            const float* __restrict__ segsum, const int* __restrict__ src,
                            const int* __restrict__ dst, float* __restrict__ hout, int nE)
{
    int e = blockIdx.x * (blockDim.x >> 5) + (threadIdx.x >> 5);
    if (e >= nE) return;
    int lane = threadIdx.x & 31;
    int c = lane * 4;
    if (e + 8 < nE)   // CDNA5 global_prefetch_b8 on the streaming operand
        __builtin_prefetch(elog + (size_t)(e + 8)*4, 0, 1);
    int s = src[e], d = dst[e];
    int head = lane >> 3;
    float alpha = elog[(size_t)e*4 + head] / (segsum[(size_t)d*4 + head] + 1e-12f);
    float4 hv = *(const float4*)(nproj + (size_t)s*384 + 256 + c);
    float* out = hout + (size_t)d*DIM + c;
    atomicAdd(out + 0, hv.x * alpha);
    atomicAdd(out + 1, hv.y * alpha);
    atomicAdd(out + 2, hv.z * alpha);
    atomicAdd(out + 3, hv.w * alpha);
}

// ----------------------------------------------------------- set2set + LSTM
__global__ void lstm_layer_k(const float* __restrict__ x, int in_dim,
                             const float* __restrict__ Wih, const float* __restrict__ Whh,
                             const float* __restrict__ bih, const float* __restrict__ bhh,
                             float* __restrict__ h, float* __restrict__ c)
{
    __shared__ float gates[512];
    int g = blockIdx.x;       // graph
    int j = threadIdx.x;      // gate index 0..511
    const float* xr = x + (size_t)g * in_dim;
    const float* hr = h + (size_t)g * DIM;
    float acc = bih[j] + bhh[j];
    const float* wi = Wih + (size_t)j * in_dim;
    for (int k = 0; k < in_dim; ++k) acc += xr[k] * wi[k];
    const float* wh = Whh + (size_t)j * DIM;
    for (int k = 0; k < DIM; ++k) acc += hr[k] * wh[k];
    gates[j] = acc;
    __syncthreads();
    if (j < DIM) {
        float ig = sigm_(gates[j]);
        float fg = sigm_(gates[DIM + j]);
        float gg = tanhf(gates[2*DIM + j]);
        float og = sigm_(gates[3*DIM + j]);
        float cn = fg * c[(size_t)g*DIM + j] + ig * gg;
        float hn = og * tanhf(cn);
        c[(size_t)g*DIM + j] = cn;
        h[(size_t)g*DIM + j] = hn;
    }
}

__global__ void s2s_dot_k(const float* __restrict__ feat, const float* __restrict__ q,
                          const int* __restrict__ gid, float* __restrict__ ev, int M)
{
    int m = blockIdx.x * (blockDim.x >> 5) + (threadIdx.x >> 5);
    if (m >= M) return;
    int lane = threadIdx.x & 31;
    int g = gid[m];
    float4 fv = *(const float4*)(feat + (size_t)m*DIM + lane*4);
    float4 qv = *(const float4*)(q + (size_t)g*DIM + lane*4);
    float p = fv.x*qv.x + fv.y*qv.y + fv.z*qv.z + fv.w*qv.w;
    for (int s = 1; s < 32; s <<= 1) p += __shfl_xor(p, s, 32);
    if (lane == 0) ev[m] = p;
}

__global__ void s2s_max_k(const float* __restrict__ ev, const int* __restrict__ gid,
                          float* __restrict__ gmax, int M){
    long i = (long)blockIdx.x*blockDim.x + threadIdx.x;
    if (i < M) atomicMaxF(&gmax[gid[i]], ev[i]);
}

__global__ void s2s_expsum_k(float* __restrict__ ev, const int* __restrict__ gid,
                             const float* __restrict__ gmax, float* __restrict__ gsum, int M){
    long i = (long)blockIdx.x*blockDim.x + threadIdx.x;
    if (i >= M) return;
    int g = gid[i];
    float ex = __expf(ev[i] - gmax[g]);
    ev[i] = ex;
    atomicAdd(&gsum[g], ex);
}

__global__ void s2s_readout_k(const float* __restrict__ feat, const float* __restrict__ ev,
                              const float* __restrict__ gsum, const int* __restrict__ gid,
                              float* __restrict__ r, int M)
{
    int m = blockIdx.x * (blockDim.x >> 5) + (threadIdx.x >> 5);
    if (m >= M) return;
    int lane = threadIdx.x & 31;
    int g = gid[m];
    float alpha = ev[m] / (gsum[g] + 1e-12f);
    float4 fv = *(const float4*)(feat + (size_t)m*DIM + lane*4);
    float* rr = r + (size_t)g*DIM + lane*4;
    atomicAdd(rr + 0, fv.x * alpha);
    atomicAdd(rr + 1, fv.y * alpha);
    atomicAdd(rr + 2, fv.z * alpha);
    atomicAdd(rr + 3, fv.w * alpha);
}

__global__ void s2s_qstar_k(const float* __restrict__ q, const float* __restrict__ r,
                            float* __restrict__ qstar){
    int g = blockIdx.x, j = threadIdx.x;                      // block 256
    qstar[(size_t)g*256 + j] = (j < DIM) ? q[(size_t)g*DIM + j]
                                         : r[(size_t)g*DIM + j - DIM];
}

__global__ void concat_vec_k(const float* __restrict__ qn, const float* __restrict__ qe,
                             float* __restrict__ vec){
    int g = blockIdx.x, j = threadIdx.x;                      // block 512
    vec[(size_t)g*512 + j] = (j < 256) ? qn[(size_t)g*256 + j]
                                       : qe[(size_t)g*256 + j - 256];
}

// ------------------------------------------------------------- gated MLP
__global__ void mlp_layer_k(const float* __restrict__ hin, const float* __restrict__ gin,
                            const float* __restrict__ W, const float* __restrict__ b,
                            const float* __restrict__ G, const float* __restrict__ gb,
                            float* __restrict__ hout, float* __restrict__ gout,
                            float* __restrict__ fin, int in_dim, int out_dim, int last)
{
    int g = blockIdx.x, j = threadIdx.x;
    if (j >= out_dim) return;
    float ha = b[j], ga = gb[j];
    const float* hr = hin + (size_t)g*in_dim;
    const float* gr = gin + (size_t)g*in_dim;
    for (int k = 0; k < in_dim; ++k){
        ha += hr[k] * W[(size_t)k*out_dim + j];
        ga += gr[k] * G[(size_t)k*out_dim + j];
    }
    if (!last){
        hout[(size_t)g*out_dim + j] = silu_(ha);
        gout[(size_t)g*out_dim + j] = silu_(ga);
    } else {
        fin[g] = ha * sigm_(ga);      // last: h linear, g sigmoid, product
    }
}

// ---------------------------------------------------------------- launcher
extern "C" void kernel_launch(void* const* d_in, const int* in_sizes, int n_in,
                              void* d_out, int out_size, void* d_ws, size_t ws_size,
                              hipStream_t stream)
{
    (void)in_sizes; (void)n_in; (void)out_size; (void)ws_size;
    const int*   nfeats   = (const int*)  d_in[0];
    const float* efeats   = (const float*)d_in[1];
    const int*   src      = (const int*)  d_in[2];
    const int*   dst      = (const int*)  d_in[3];
    const int*   node_gid = (const int*)  d_in[4];
    const int*   edge_gid = (const int*)  d_in[5];
    const float* embed    = (const float*)d_in[6];
    struct ConvP { const float *Wni,*Wnj,*Wfij,*b,*attn,*Wnode; };
    ConvP conv[2];
    for (int l = 0; l < 2; ++l){
        int o = 7 + l*6;
        conv[l] = { (const float*)d_in[o+0], (const float*)d_in[o+1],
                    (const float*)d_in[o+2], (const float*)d_in[o+3],
                    (const float*)d_in[o+4], (const float*)d_in[o+5] };
    }
    const float* nWih  = (const float*)d_in[19];
    const float* nWhh  = (const float*)d_in[20];
    const float* nbih  = (const float*)d_in[21];
    const float* nbhh  = (const float*)d_in[22];
    const float* eWih0 = (const float*)d_in[23];
    const float* eWih1 = (const float*)d_in[24];
    const float* eWhh0 = (const float*)d_in[25];
    const float* eWhh1 = (const float*)d_in[26];
    const float* ebih0 = (const float*)d_in[27];
    const float* ebih1 = (const float*)d_in[28];
    const float* ebhh0 = (const float*)d_in[29];
    const float* ebhh1 = (const float*)d_in[30];
    const float* Ws0 = (const float*)d_in[31];
    const float* Ws1 = (const float*)d_in[32];
    const float* Ws2 = (const float*)d_in[33];
    const float* bs0 = (const float*)d_in[34];
    const float* bs1 = (const float*)d_in[35];
    const float* bs2 = (const float*)d_in[36];
    const float* Gs0 = (const float*)d_in[37];
    const float* Gs1 = (const float*)d_in[38];
    const float* Gs2 = (const float*)d_in[39];
    const float* gb0 = (const float*)d_in[40];
    const float* gb1 = (const float*)d_in[41];
    const float* gb2 = (const float*)d_in[42];

    // ---- workspace carve (bytes, 256B aligned)
    char* w = (char*)d_ws;
    auto carveB = [&](size_t bytes){ void* p = (void*)w;
        w += (bytes + 255) & ~(size_t)255; return p; };
    auto carve = [&](size_t elems){ return (float*)carveB(elems*sizeof(float)); };
    float*    NF0   = carve((size_t)NNODES*DIM);
    float*    NF1   = carve((size_t)NNODES*DIM);
    float*    NPROJ = carve((size_t)NNODES*384);
    float*    FA    = carve((size_t)NEDGES*DIM);
    float*    FB    = carve((size_t)NEDGES*DIM);
    float*    ELOG  = carve((size_t)NEDGES*4);    // also reused as set2set e-vector
    float*    SMAX  = carve((size_t)NNODES*4);
    float*    SSUM  = carve((size_t)NNODES*4);
    _Float16* WH    = (_Float16*)carveB(16384*sizeof(_Float16));  // packed f16 weight
    float*    HS0   = carve(NGRAPH*DIM);
    float*    CS0   = carve(NGRAPH*DIM);
    float*    EH0   = carve(NGRAPH*DIM);
    float*    EC0   = carve(NGRAPH*DIM);
    float*    EH1   = carve(NGRAPH*DIM);
    float*    EC1   = carve(NGRAPH*DIM);
    float*    QSN   = carve(NGRAPH*256);
    float*    QSE   = carve(NGRAPH*256);
    float*    RBUF  = carve(NGRAPH*DIM);
    float*    GMAX  = carve(NGRAPH);
    float*    GSUM  = carve(NGRAPH);
    float*    VEC   = carve(NGRAPH*512);
    float*    H1    = carve(NGRAPH*256);
    float*    G1    = carve(NGRAPH*256);
    float*    H2    = carve(NGRAPH*DIM);
    float*    G2    = carve(NGRAPH*DIM);

    auto blk1d = [](long n){ return dim3((unsigned)((n + 255) / 256)); };
    const dim3 B256(256);

    // node GEMM helper: pack weight into B-fragment order, then WMMA GEMM
    auto gemm = [&](const float* A, int lda, const float* W32,
                    float* C, int ldc, int ccol, int M){
        pack_b_k<<<dim3(64), B256, 0, stream>>>(W32, WH);
        gemm128_wmma_f16<<<dim3(M/16), B256, 0, stream>>>(A, lda, WH, C, ldc, ccol);
    };

    // ---- embedding
    embed_k<<<blk1d((long)NNODES*DIM), B256, 0, stream>>>(nfeats, embed, NF0, (long)NNODES*DIM);

    // ---- two EGAT conv layers
    for (int l = 0; l < 2; ++l){
        const float* nfin  = (l == 0) ? NF0 : NF1;
        float*       nfout = (l == 0) ? NF1 : NF0;
        const float* efin  = (l == 0) ? efeats : FA;
        float*       F     = (l == 0) ? FA : FB;

        // node projections packed into one L2-resident N x 384 table (76.8 MB < 192 MB L2)
        gemm(nfin, DIM, conv[l].Wni,   NPROJ, 384,   0, NNODES);
        gemm(nfin, DIM, conv[l].Wnj,   NPROJ, 384, 128, NNODES);
        gemm(nfin, DIM, conv[l].Wnode, NPROJ, 384, 256, NNODES);

        // fused: ef @ W_fij + gathers + bias + lrelu + head logits + elu, one pass
        pack_b_k<<<dim3(64), B256, 0, stream>>>(conv[l].Wfij, WH);
        gemm_edge_fused<<<dim3(NEDGES/16), B256, 0, stream>>>(
            efin, WH, NPROJ, conv[l].b, conv[l].attn, src, dst, F, ELOG);

        fill_k<<<blk1d((long)NNODES*4), B256, 0, stream>>>(SMAX, -INFINITY, (long)NNODES*4);
        fill_k<<<blk1d((long)NNODES*4), B256, 0, stream>>>(SSUM, 0.f, (long)NNODES*4);
        seg_max_k   <<<blk1d((long)NEDGES*4), B256, 0, stream>>>(ELOG, dst, SMAX, NEDGES);
        seg_expsum_k<<<blk1d((long)NEDGES*4), B256, 0, stream>>>(ELOG, dst, SMAX, SSUM, NEDGES);

        fill_k<<<blk1d((long)NNODES*DIM), B256, 0, stream>>>(nfout, 0.f, (long)NNODES*DIM);
        aggregate_k<<<dim3(NEDGES/8), B256, 0, stream>>>(NPROJ, ELOG, SSUM, src, dst, nfout, NEDGES);
        elu_k<<<blk1d((long)NNODES*DIM), B256, 0, stream>>>(nfout, (long)NNODES*DIM);
        // edge features F already hold elu(f) from the fused epilogue
    }
    // final node feats: NF0, final edge feats: FB

    // ---- node set2set (1-layer LSTM, 3 iterations)
    fill_k<<<blk1d(NGRAPH*DIM), B256, 0, stream>>>(HS0, 0.f, NGRAPH*DIM);
    fill_k<<<blk1d(NGRAPH*DIM), B256, 0, stream>>>(CS0, 0.f, NGRAPH*DIM);
    fill_k<<<blk1d(NGRAPH*256), B256, 0, stream>>>(QSN, 0.f, NGRAPH*256);
    for (int it = 0; it < 3; ++it){
        lstm_layer_k<<<dim3(NGRAPH), dim3(512), 0, stream>>>(QSN, 256, nWih, nWhh, nbih, nbhh, HS0, CS0);
        s2s_dot_k<<<dim3(NNODES/8 + 1), B256, 0, stream>>>(NF0, HS0, node_gid, ELOG, NNODES);
        fill_k<<<dim3(1), dim3(32), 0, stream>>>(GMAX, -INFINITY, NGRAPH);
        fill_k<<<dim3(1), dim3(32), 0, stream>>>(GSUM, 0.f, NGRAPH);
        s2s_max_k   <<<blk1d(NNODES), B256, 0, stream>>>(ELOG, node_gid, GMAX, NNODES);
        s2s_expsum_k<<<blk1d(NNODES), B256, 0, stream>>>(ELOG, node_gid, GMAX, GSUM, NNODES);
        fill_k<<<blk1d(NGRAPH*DIM), B256, 0, stream>>>(RBUF, 0.f, NGRAPH*DIM);
        s2s_readout_k<<<dim3(NNODES/8 + 1), B256, 0, stream>>>(NF0, ELOG, GSUM, node_gid, RBUF, NNODES);
        s2s_qstar_k<<<dim3(NGRAPH), B256, 0, stream>>>(HS0, RBUF, QSN);
    }

    // ---- edge set2set (2-layer LSTM, 1 iteration)
    fill_k<<<blk1d(NGRAPH*DIM), B256, 0, stream>>>(EH0, 0.f, NGRAPH*DIM);
    fill_k<<<blk1d(NGRAPH*DIM), B256, 0, stream>>>(EC0, 0.f, NGRAPH*DIM);
    fill_k<<<blk1d(NGRAPH*DIM), B256, 0, stream>>>(EH1, 0.f, NGRAPH*DIM);
    fill_k<<<blk1d(NGRAPH*DIM), B256, 0, stream>>>(EC1, 0.f, NGRAPH*DIM);
    fill_k<<<blk1d(NGRAPH*256), B256, 0, stream>>>(QSE, 0.f, NGRAPH*256);
    {
        lstm_layer_k<<<dim3(NGRAPH), dim3(512), 0, stream>>>(QSE, 256, eWih0, eWhh0, ebih0, ebhh0, EH0, EC0);
        lstm_layer_k<<<dim3(NGRAPH), dim3(512), 0, stream>>>(EH0, 128, eWih1, eWhh1, ebih1, ebhh1, EH1, EC1);
        s2s_dot_k<<<dim3(NEDGES/8), B256, 0, stream>>>(FB, EH1, edge_gid, ELOG, NEDGES);
        fill_k<<<dim3(1), dim3(32), 0, stream>>>(GMAX, -INFINITY, NGRAPH);
        fill_k<<<dim3(1), dim3(32), 0, stream>>>(GSUM, 0.f, NGRAPH);
        s2s_max_k   <<<blk1d(NEDGES), B256, 0, stream>>>(ELOG, edge_gid, GMAX, NEDGES);
        s2s_expsum_k<<<blk1d(NEDGES), B256, 0, stream>>>(ELOG, edge_gid, GMAX, GSUM, NEDGES);
        fill_k<<<blk1d(NGRAPH*DIM), B256, 0, stream>>>(RBUF, 0.f, NGRAPH*DIM);
        s2s_readout_k<<<dim3(NEDGES/8), B256, 0, stream>>>(FB, ELOG, GSUM, edge_gid, RBUF, NEDGES);
        s2s_qstar_k<<<dim3(NGRAPH), B256, 0, stream>>>(EH1, RBUF, QSE);
    }

    // ---- gated MLP head: 512 -> 256 -> 128 -> 1
    concat_vec_k<<<dim3(NGRAPH), dim3(512), 0, stream>>>(QSN, QSE, VEC);
    mlp_layer_k<<<dim3(NGRAPH), dim3(256), 0, stream>>>(VEC, VEC, Ws0, bs0, Gs0, gb0, H1, G1, nullptr, 512, 256, 0);
    mlp_layer_k<<<dim3(NGRAPH), dim3(128), 0, stream>>>(H1, G1, Ws1, bs1, Gs1, gb1, H2, G2, nullptr, 256, 128, 0);
    mlp_layer_k<<<dim3(NGRAPH), dim3(1),   0, stream>>>(H2, G2, Ws2, bs2, Gs2, gb2, nullptr, nullptr,
                                                        (float*)d_out, 128, 1, 1);
}
